// MultiHeadAttention_38182259261891
// MI455X (gfx1250) — matrix operand order repaired and checked by reading, not compiled, not run on hardware
//
#include <hip/hip_runtime.h>

// ---------------- MHA fwd: B=2, S=2048, H=1024, NH=16, HD=64 ----------------
#define Bsz 2
#define Ssz 2048
#define Hsz 1024
#define NHd 16
#define HDd 64
#define Mrows (Bsz * Ssz)        // 4096
#define QKVN (3 * Hsz)           // 3072

typedef __attribute__((ext_vector_type(16))) __bf16 bf16x16;
typedef __attribute__((ext_vector_type(8)))  __bf16 bf16x8;
typedef __attribute__((ext_vector_type(8)))  float  f32x8;

__device__ __forceinline__ f32x8 wmma_bf16(bf16x16 a, bf16x16 b, f32x8 c) {
  return __builtin_amdgcn_wmma_f32_16x16x32_bf16(false, a, false, b,
                                                 (short)0, c, false, false);
}

// A-matrix 16x32 bf16 fragment: lane L holds row M=L%16,
// contiguous K [base..base+15], base=(L/16)*16.
__device__ __forceinline__ bf16x16 load_a_frag(const __bf16* p) {
  bf16x8 lo = *(const bf16x8*)p;
  bf16x8 hi = *(const bf16x8*)(p + 8);
  bf16x16 a;
#pragma unroll
  for (int i = 0; i < 8; ++i) { a[i] = lo[i]; a[i + 8] = hi[i]; }
  return a;
}

// ------------------------------ fp32 -> bf16 --------------------------------
__global__ void __launch_bounds__(256)
cvt_f32_bf16(const float* __restrict__ src, __bf16* __restrict__ dst, int n) {
  for (int i = blockIdx.x * blockDim.x + threadIdx.x; i < n;
       i += gridDim.x * blockDim.x)
    dst[i] = (__bf16)src[i];
}

// ------------------- QKV projection GEMM + scatter to Q/K/V^T ---------------
// C[m,n] = sum_k X[m,k] * Wqkv[n,k];  M=4096, N=3072, K=1024
// grid = (48, 32), block = 256 (8 waves); wave -> 16(M) x 64(N) strip.
__global__ void __launch_bounds__(256)
gemm_qkv(const __bf16* __restrict__ X, const __bf16* __restrict__ W,
         __bf16* __restrict__ Qb, __bf16* __restrict__ Kb,
         __bf16* __restrict__ Vt) {
  const int lane = threadIdx.x & 31;
  const int wave = threadIdx.x >> 5;
  const int m0 = (blockIdx.y * 8 + wave) * 16;
  const int nb = blockIdx.x * 64;

  const __bf16* arow =
      X + (size_t)(m0 + (lane & 15)) * Hsz + ((lane >> 4) << 4);
  f32x8 acc[4] = {};
#pragma unroll 4
  for (int k = 0; k < Hsz; k += 32) {
    __builtin_prefetch(arow + k + 128, 0, 0);
    bf16x16 a = load_a_frag(arow + k);
#pragma unroll
    for (int t = 0; t < 4; ++t) {
      // B fragment: lane holds col n=lane%16 (a W row), 16 contiguous k.
      const __bf16* brow =
          W + (size_t)(nb + t * 16 + (lane & 15)) * Hsz + ((lane >> 4) << 4) + k;
      acc[t] = wmma_bf16(a, *(const bf16x16*)brow, acc[t]);
    }
  }
  // Epilogue scatter: C layout -> lane has n=nb+t*16+lane%16, rows m=r+8*(lane/16)
#pragma unroll
  for (int t = 0; t < 4; ++t) {
    const int n = nb + t * 16 + (lane & 15);
    const int sec = n >> 10, f = n & 1023, head = f >> 6, d = f & 63;
#pragma unroll
    for (int r = 0; r < 8; ++r) {
      const int m = m0 + r + ((lane >> 4) << 3);
      const int b = m >> 11, s = m & 2047;
      const int bh = b * NHd + head;
      const __bf16 v = (__bf16)acc[t][r];
      if (sec == 0)      Qb[((size_t)bh * Ssz + s) * HDd + d] = v;
      else if (sec == 1) Kb[((size_t)bh * Ssz + s) * HDd + d] = v;
      else               Vt[((size_t)bh * HDd + d) * Ssz + s] = v;  // transposed
    }
  }
}

// ----------------------------------- RoPE -----------------------------------
__global__ void __launch_bounds__(256)
rope_kernel(__bf16* __restrict__ Qb, __bf16* __restrict__ Kb) {
  const int idx = blockIdx.x * blockDim.x + threadIdx.x; // B*NH*S*32 threads
  const int d = idx & 31;
  const int s = (idx >> 5) & (Ssz - 1);
  const int bh = idx >> 16;
  const float invf = __powf(10000.f, -(float)d / 32.f);
  float sn, cs;
  __sincosf((float)s * invf, &sn, &cs);
  const size_t base = ((size_t)bh * Ssz + s) * HDd + d;
  {
    float x1 = (float)Qb[base], x2 = (float)Qb[base + 32];
    Qb[base]      = (__bf16)(x1 * cs - x2 * sn);
    Qb[base + 32] = (__bf16)(x2 * cs + x1 * sn);
  }
  {
    float x1 = (float)Kb[base], x2 = (float)Kb[base + 32];
    Kb[base]      = (__bf16)(x1 * cs - x2 * sn);
    Kb[base + 32] = (__bf16)(x2 * cs + x1 * sn);
  }
}

// ------------------------- Flash attention (online softmax) -----------------
// grid = B*NH*(S/64) = 1024 blocks, block = 128 (4 waves), wave -> 16 q rows.
// 64 k-columns per softmax window: 8 score WMMAs + 8 PV WMMAs per iteration,
// halving shuffle-reduction / rescale cost per column vs a 32-col window.
__global__ void __launch_bounds__(128)
attn_kernel(const __bf16* __restrict__ Qb, const __bf16* __restrict__ Kb,
            const __bf16* __restrict__ Vt, __bf16* __restrict__ Ob) {
  __shared__ __align__(32) __bf16 Psh[4][16][64];  // per-wave P tile (C->A relayout)
  const int lane = threadIdx.x & 31;
  const int wave = threadIdx.x >> 5;
  const int bh = blockIdx.x >> 5;                  // b*16 + head
  const int q0 = (blockIdx.x & 31) * 64 + wave * 16;

  const __bf16* Qh = Qb + (size_t)bh * Ssz * HDd;
  const __bf16* Kh = Kb + (size_t)bh * Ssz * HDd;
  const __bf16* Vh = Vt + (size_t)bh * HDd * Ssz;  // [d][s]

  const __bf16* qrow = Qh + (size_t)(q0 + (lane & 15)) * HDd + ((lane >> 4) << 4);
  const bf16x16 aq0 = load_a_frag(qrow);       // d 0..31
  const bf16x16 aq1 = load_a_frag(qrow + 32);  // d 32..63

  float mrow[8], lrow[8];
  f32x8 acc[4] = {};
#pragma unroll
  for (int r = 0; r < 8; ++r) { mrow[r] = -1e30f; lrow[r] = 0.f; }

  const int half16 = (lane >> 4) << 4;
  for (int k0 = 0; k0 < Ssz; k0 += 64) {
    // ---- scores: four 16x16 tiles, K-dim = HD = 64 (2 WMMAs each) ----
    f32x8 sc[4];
#pragma unroll
    for (int t2 = 0; t2 < 4; ++t2) {
      const __bf16* kr =
          Kh + (size_t)(k0 + t2 * 16 + (lane & 15)) * HDd + half16;
      __builtin_prefetch(kr + 64 * HDd, 0, 0);
      f32x8 s = {};
      s = wmma_bf16(aq0, *(const bf16x16*)kr, s);
      s = wmma_bf16(aq1, *(const bf16x16*)(kr + 32), s);
      sc[t2] = s;
    }

    // ---- online softmax over 64 columns; row m lives across 16 lanes ----
#pragma unroll
    for (int r = 0; r < 8; ++r) {
      float a0 = sc[0][r] * 0.125f, a1 = sc[1][r] * 0.125f;   // HD^-0.5
      float a2 = sc[2][r] * 0.125f, a3 = sc[3][r] * 0.125f;
      float mx = fmaxf(fmaxf(a0, a1), fmaxf(a2, a3));
#pragma unroll
      for (int off = 8; off >= 1; off >>= 1)
        mx = fmaxf(mx, __shfl_xor(mx, off, 16));
      const float newm = fmaxf(mrow[r], mx);
      const float alpha = __expf(mrow[r] - newm);
      const float p0 = __expf(a0 - newm);
      const float p1 = __expf(a1 - newm);
      const float p2 = __expf(a2 - newm);
      const float p3 = __expf(a3 - newm);
      float rs = (p0 + p1) + (p2 + p3);
#pragma unroll
      for (int off = 8; off >= 1; off >>= 1)
        rs += __shfl_xor(rs, off, 16);
      lrow[r] = lrow[r] * alpha + rs;
      mrow[r] = newm;
#pragma unroll
      for (int t = 0; t < 4; ++t) acc[t][r] *= alpha;
      const int m = r + ((lane >> 4) << 3);
      const int n = lane & 15;
      Psh[wave][m][n]      = (__bf16)p0;
      Psh[wave][m][16 + n] = (__bf16)p1;
      Psh[wave][m][32 + n] = (__bf16)p2;
      Psh[wave][m][48 + n] = (__bf16)p3;
    }
    asm volatile("s_wait_dscnt 0" ::: "memory");

    // ---- P(16x64) x V(64x64): 2 A-frags from LDS, B-frags from V^T ----
    const bf16x16 ap0 =
        load_a_frag(&Psh[wave][lane & 15][(lane >> 4) << 4]);
    const bf16x16 ap1 =
        load_a_frag(&Psh[wave][lane & 15][32 + ((lane >> 4) << 4)]);
#pragma unroll
    for (int t = 0; t < 4; ++t) {
      const __bf16* vr =
          Vh + (size_t)(t * 16 + (lane & 15)) * Ssz + k0 + half16;
      acc[t] = wmma_bf16(ap0, *(const bf16x16*)vr, acc[t]);
      acc[t] = wmma_bf16(ap1, *(const bf16x16*)(vr + 32), acc[t]);
    }
  }

  // ---- normalize + store to (B,S,H) bf16 ----
  const int b = bh >> 4, head = bh & 15;
#pragma unroll
  for (int t = 0; t < 4; ++t) {
    const int col = head * HDd + t * 16 + (lane & 15);
#pragma unroll
    for (int r = 0; r < 8; ++r) {
      const int s = q0 + r + ((lane >> 4) << 3);
      Ob[((size_t)b * Ssz + s) * Hsz + col] = (__bf16)(acc[t][r] / lrow[r]);
    }
  }
}

// --------------------------- Output projection GEMM -------------------------
// out[m,n] = sum_k A[m,k] * Wout[n,k];  M=4096, N=1024, K=1024 -> fp32 out.
__global__ void __launch_bounds__(256)
gemm_out(const __bf16* __restrict__ A, const __bf16* __restrict__ W,
         float* __restrict__ out) {
  const int lane = threadIdx.x & 31;
  const int wave = threadIdx.x >> 5;
  const int m0 = (blockIdx.y * 8 + wave) * 16;
  const int nb = blockIdx.x * 64;

  const __bf16* arow =
      A + (size_t)(m0 + (lane & 15)) * Hsz + ((lane >> 4) << 4);
  f32x8 acc[4] = {};
#pragma unroll 4
  for (int k = 0; k < Hsz; k += 32) {
    __builtin_prefetch(arow + k + 128, 0, 0);
    bf16x16 a = load_a_frag(arow + k);
#pragma unroll
    for (int t = 0; t < 4; ++t) {
      const __bf16* brow =
          W + (size_t)(nb + t * 16 + (lane & 15)) * Hsz + ((lane >> 4) << 4) + k;
      acc[t] = wmma_bf16(a, *(const bf16x16*)brow, acc[t]);
    }
  }
#pragma unroll
  for (int t = 0; t < 4; ++t) {
    const int n = nb + t * 16 + (lane & 15);
#pragma unroll
    for (int r = 0; r < 8; ++r) {
      const int m = m0 + r + ((lane >> 4) << 3);
      out[(size_t)m * Hsz + n] = acc[t][r];
    }
  }
}

// --------------------------------- launcher ---------------------------------
extern "C" void kernel_launch(void* const* d_in, const int* in_sizes, int n_in,
                              void* d_out, int out_size, void* d_ws,
                              size_t ws_size, hipStream_t stream) {
  const float* x    = (const float*)d_in[0];   // (B,S,H)
  const float* wqkv = (const float*)d_in[1];   // (3H,H)
  const float* wout = (const float*)d_in[2];   // (H,H)
  float* out = (float*)d_out;

  char* p = (char*)d_ws;
  __bf16* xb    = (__bf16*)p; p += (size_t)Mrows * Hsz * 2;      // 8 MiB
  __bf16* wqkvb = (__bf16*)p; p += (size_t)QKVN * Hsz * 2;       // 6 MiB
  __bf16* woutb = (__bf16*)p; p += (size_t)Hsz * Hsz * 2;        // 2 MiB
  __bf16* Qb    = (__bf16*)p; p += (size_t)Bsz * NHd * Ssz * HDd * 2;
  __bf16* Kb    = (__bf16*)p; p += (size_t)Bsz * NHd * Ssz * HDd * 2;
  __bf16* Vt    = (__bf16*)p; p += (size_t)Bsz * NHd * HDd * Ssz * 2;
  __bf16* Ab    = (__bf16*)p; p += (size_t)Mrows * Hsz * 2;      // attn out

  cvt_f32_bf16<<<4096, 256, 0, stream>>>(x, xb, Mrows * Hsz);
  cvt_f32_bf16<<<4096, 256, 0, stream>>>(wqkv, wqkvb, QKVN * Hsz);
  cvt_f32_bf16<<<2048, 256, 0, stream>>>(wout, woutb, Hsz * Hsz);

  gemm_qkv<<<dim3(QKVN / 64, Mrows / 128), 256, 0, stream>>>(xb, wqkvb, Qb, Kb, Vt);
  rope_kernel<<<(Bsz * NHd * Ssz * 32) / 256, 256, 0, stream>>>(Qb, Kb);
  attn_kernel<<<Bsz * NHd * (Ssz / 64), 128, 0, stream>>>(Qb, Kb, Vt, Ab);
  gemm_out<<<dim3(Hsz / 64, Mrows / 128), 256, 0, stream>>>(Ab, woutb, out);
}
